// ByteContextEncoder_86921548137294
// MI455X (gfx1250) — compile-verified, matrix operand order checked by reading, not run on hardware
//
#include <hip/hip_runtime.h>

#define B_  4
#define T_  2048
#define D_  512
#define H_  8
#define L_  2
#define FF_ 2048
#define HD_ 64
#define BT_ (B_ * T_)
#define NSEG_ (B_ * (T_ + 1))

typedef __bf16 v16bf __attribute__((ext_vector_type(16)));
typedef float  v8f   __attribute__((ext_vector_type(8)));

#define WMMA_BF16(a, b, c) \
  __builtin_amdgcn_wmma_f32_16x16x32_bf16(false, (a), false, (b), (short)0, (c), false, false)

// ---------------------------------------------------------------------------
// helpers
// ---------------------------------------------------------------------------
__device__ __forceinline__ __bf16 f2bf(float f) {
  unsigned u = __builtin_bit_cast(unsigned, f);
  unsigned r = (u + 0x7FFFu + ((u >> 16) & 1u)) >> 16;
  unsigned short s = (unsigned short)r;
  return __builtin_bit_cast(__bf16, s);
}

// Load a 16x32 bf16 WMMA operand fragment (A layout; B symmetric with
// "row" = output column) from row-major storage, row stride ldk elements.
// ISA 7.12.2: lanes 0-15 hold K {0..7,16..23}, lanes 16-31 hold K
// {8..15,24..31}; consecutive K pairs share a dword -> 8 dword loads,
// fused by the backend into two global_load_b128.
__device__ __forceinline__ v16bf load_frag16(const __bf16* base, int row0,
                                             int ldk, int k0, int lane) {
  const int r  = lane & 15;
  const int hf = (lane >> 4) & 1;
  const unsigned* p32 =
      (const unsigned*)(base + (size_t)(row0 + r) * ldk + k0 + hf * 8);
  union { v16bf v; unsigned u[8]; } f;
#pragma unroll
  for (int i = 0; i < 4; ++i) f.u[i] = p32[i];          // K = hf*8 + {0,2,4,6}
#pragma unroll
  for (int i = 0; i < 4; ++i) f.u[4 + i] = p32[8 + i];  // K = 16 + hf*8 + ...
  return f.v;
}

// ---------------------------------------------------------------------------
// generic bf16 WMMA GEMM:  C[M,N] = A[M,K] @ Wt[N,K]^T  (Wt = pre-transposed)
// K unrolled by 2 with ping-pong fragment buffers: loads for step i+1 issue
// before the WMMAs of step i, and each buffer is rewritten in place by its
// own loads -> no register-rotation moves, staged s_wait_loadcnt.
// Requires K % 64 == 0 (all call sites: 512 or 2048).
// mode 0: outF = acc
// mode 1: outF = aux + acc                 (residual add)
// mode 2: outB = bf16( silu(aux) * acc )   (SwiGLU gate fusion)
// ---------------------------------------------------------------------------
__global__ __launch_bounds__(256) void gemm_bf16_wmma(
    const __bf16* __restrict__ A, const __bf16* __restrict__ Wt,
    float* __restrict__ outF, __bf16* __restrict__ outB,
    const float* __restrict__ aux, int M, int N, int K, int mode) {
  const int lane = threadIdx.x & 31;
  const int wave = blockIdx.x * (blockDim.x >> 5) + (threadIdx.x >> 5);
  const int ntiles = N >> 6;                  // 64 columns per wave
  const int mt = wave / ntiles;
  const int nt = wave % ntiles;
  if (mt * 16 >= M) return;

  v8f acc[4];
#pragma unroll
  for (int t = 0; t < 4; ++t) acc[t] = (v8f){0, 0, 0, 0, 0, 0, 0, 0};

  v16bf a0 = load_frag16(A, mt * 16, K, 0, lane);
  v16bf b0[4];
#pragma unroll
  for (int t = 0; t < 4; ++t)
    b0[t] = load_frag16(Wt, nt * 64 + t * 16, K, 0, lane);

  for (int k0 = 0; k0 < K; k0 += 64) {
    __builtin_prefetch(A + (size_t)(mt * 16 + (lane & 15)) * K + k0 + 64, 0, 1);
    // stage k0+32 into the odd buffers while even-buffer WMMAs run
    v16bf a1 = load_frag16(A, mt * 16, K, k0 + 32, lane);
    v16bf b1[4];
#pragma unroll
    for (int t = 0; t < 4; ++t)
      b1[t] = load_frag16(Wt, nt * 64 + t * 16, K, k0 + 32, lane);
#pragma unroll
    for (int t = 0; t < 4; ++t) acc[t] = WMMA_BF16(a0, b0[t], acc[t]);
    // stage k0+64 back into the even buffers while odd-buffer WMMAs run
    if (k0 + 64 < K) {
      a0 = load_frag16(A, mt * 16, K, k0 + 64, lane);
#pragma unroll
      for (int t = 0; t < 4; ++t)
        b0[t] = load_frag16(Wt, nt * 64 + t * 16, K, k0 + 64, lane);
    }
#pragma unroll
    for (int t = 0; t < 4; ++t) acc[t] = WMMA_BF16(a1, b1[t], acc[t]);
  }

  const int r0 = lane & 15;
  const int hf = (lane >> 4) & 1;
#pragma unroll
  for (int t = 0; t < 4; ++t) {
    const int col = nt * 64 + t * 16 + r0;
#pragma unroll
    for (int r = 0; r < 8; ++r) {
      const int m = mt * 16 + r + 8 * hf;
      const size_t idx = (size_t)m * N + col;
      const float v = acc[t][r];
      if (mode == 1) {
        outF[idx] = aux[idx] + v;
      } else if (mode == 2) {
        const float u = aux[idx];
        const float s = u / (1.0f + __expf(-u));
        outB[idx] = f2bf(s * v);
      } else {
        outF[idx] = v;
      }
    }
  }
}

// ---------------------------------------------------------------------------
// flash attention (full, non-causal): 4 waves/block share one (b,h); each
// wave owns a 16-query tile. Q pre-scaled by 1/sqrt(HD). The 32x64 V tile is
// staged cooperatively into LDS with coalesced b64 loads; P and V fragments
// are then gathered from LDS.
// ---------------------------------------------------------------------------
#define ATT_WAVES 4
#define VPAD 8
__global__ __launch_bounds__(ATT_WAVES * 32) void attn_wmma(
    const __bf16* __restrict__ Qb, const __bf16* __restrict__ Kb,
    const __bf16* __restrict__ Vb, __bf16* __restrict__ Ob) {
  __shared__ __bf16 ptile[ATT_WAVES][16][32];
  __shared__ __bf16 vtile[32][64 + VPAD];

  const int tid = threadIdx.x;
  const int lane = tid & 31;
  const int w = tid >> 5;
  const int bh = blockIdx.x;
  const int b = bh / H_;
  const int h = bh % H_;
  const int q0 = blockIdx.y * (ATT_WAVES * 16) + w * 16;

  const __bf16* Qh = Qb + (size_t)b * T_ * D_ + h * HD_;
  const __bf16* Kh = Kb + (size_t)b * T_ * D_ + h * HD_;
  const __bf16* Vh = Vb + (size_t)b * T_ * D_ + h * HD_;

  const v16bf qa0 = load_frag16(Qh, q0, D_, 0, lane);
  const v16bf qa1 = load_frag16(Qh, q0, D_, 32, lane);

  v8f o[4];
#pragma unroll
  for (int t = 0; t < 4; ++t) o[t] = (v8f){0, 0, 0, 0, 0, 0, 0, 0};
  float mi[8], li[8];
#pragma unroll
  for (int r = 0; r < 8; ++r) { mi[r] = -1e30f; li[r] = 0.0f; }

  const int r0 = lane & 15;
  const int hf = (lane >> 4) & 1;
  const int vr = tid >> 2;         // 0..31 : V tile row handled by this thread
  const int vc = (tid & 3) * 16;   // 0,16,32,48 : 16-element column chunk

  for (int j0 = 0; j0 < T_; j0 += 32) {
    // ---- S = Q K^T for 32 keys (K fragments straight from global) ----
    v8f s0 = (v8f){0, 0, 0, 0, 0, 0, 0, 0};
    v8f s1 = (v8f){0, 0, 0, 0, 0, 0, 0, 0};
    s0 = WMMA_BF16(qa0, load_frag16(Kh, j0, D_, 0, lane), s0);
    s0 = WMMA_BF16(qa1, load_frag16(Kh, j0, D_, 32, lane), s0);
    s1 = WMMA_BF16(qa0, load_frag16(Kh, j0 + 16, D_, 0, lane), s1);
    s1 = WMMA_BF16(qa1, load_frag16(Kh, j0 + 16, D_, 32, lane), s1);

    // ---- online softmax (register-resident, cross-lane reductions) ----
    float p0[8], p1[8];
#pragma unroll
    for (int r = 0; r < 8; ++r) {
      float rm = fmaxf(s0[r], s1[r]);
#pragma unroll
      for (int msk = 1; msk < 16; msk <<= 1)
        rm = fmaxf(rm, __shfl_xor(rm, msk, 32));
      const float mn = fmaxf(mi[r], rm);
      const float corr = __expf(mi[r] - mn);
      mi[r] = mn;
      const float e0 = __expf(s0[r] - mn);
      const float e1 = __expf(s1[r] - mn);
      float rs = e0 + e1;
#pragma unroll
      for (int msk = 1; msk < 16; msk <<= 1) rs += __shfl_xor(rs, msk, 32);
      li[r] = li[r] * corr + rs;
      p0[r] = e0;
      p1[r] = e1;
#pragma unroll
      for (int t = 0; t < 4; ++t) o[t][r] *= corr;
    }

    __syncthreads();  // previous iteration's LDS reads are done

    // ---- P tile: C-layout -> LDS 16x32 ----
#pragma unroll
    for (int r = 0; r < 8; ++r) {
      ptile[w][r + 8 * hf][r0] = f2bf(p0[r]);
      ptile[w][r + 8 * hf][16 + r0] = f2bf(p1[r]);
    }

    // ---- V tile: cooperative coalesced stage (32 keys x 64 dims) ----
    {
      const uint2* src = (const uint2*)(Vh + (size_t)(j0 + vr) * D_ + vc);
      uint2* dst = (uint2*)&vtile[vr][vc];
#pragma unroll
      for (int i = 0; i < 4; ++i) dst[i] = src[i];
    }

    __syncthreads();

    // ---- gather P as A fragment ----
    union { v16bf v; unsigned short us[16]; } pa;
#pragma unroll
    for (int t = 0; t < 16; ++t) {
      const int k = (t < 8 ? t : t + 8) + hf * 8;
      pa.us[t] = __builtin_bit_cast(unsigned short, ptile[w][r0][k]);
    }

    // ---- O += P V, V B-fragments from LDS ----
#pragma unroll
    for (int dt = 0; dt < 4; ++dt) {
      union { v16bf v; unsigned short us[16]; } vbf;
#pragma unroll
      for (int t = 0; t < 16; ++t) {
        const int k = (t < 8 ? t : t + 8) + hf * 8;
        vbf.us[t] = __builtin_bit_cast(unsigned short, vtile[k][dt * 16 + r0]);
      }
      o[dt] = WMMA_BF16(pa.v, vbf.v, o[dt]);
    }
  }

#pragma unroll
  for (int dt = 0; dt < 4; ++dt)
#pragma unroll
    for (int r = 0; r < 8; ++r) {
      const int m = q0 + r + 8 * hf;
      Ob[((size_t)b * T_ + m) * D_ + h * HD_ + dt * 16 + r0] =
          f2bf(o[dt][r] / li[r]);
    }
}

// ---------------------------------------------------------------------------
// elementwise / reduction kernels
// ---------------------------------------------------------------------------
__global__ void embed_k(const int* __restrict__ tok,
                        const float* __restrict__ tbl, float* __restrict__ x) {
  const int bt = blockIdx.x;
  const int tid = threadIdx.x;
  int c = tok[bt];
  c = min(max(c, 0), 255);
  x[(size_t)bt * D_ + tid] = tbl[(size_t)c * D_ + tid];
  x[(size_t)bt * D_ + tid + 256] = tbl[(size_t)c * D_ + tid + 256];
}

__global__ __launch_bounds__(256) void rmsnorm_k(const float* __restrict__ x,
                                                 const float* __restrict__ w,
                                                 __bf16* __restrict__ outb,
                                                 float* __restrict__ outf) {
  __shared__ float red[8];
  const int bt = blockIdx.x;
  const int tid = threadIdx.x;
  const float* row = x + (size_t)bt * D_;
  const float a = row[tid], b = row[tid + 256];
  float ss = a * a + b * b;
#pragma unroll
  for (int m = 1; m < 32; m <<= 1) ss += __shfl_xor(ss, m, 32);
  if ((tid & 31) == 0) red[tid >> 5] = ss;
  __syncthreads();
  float tot = 0.0f;
#pragma unroll
  for (int i = 0; i < 8; ++i) tot += red[i];
  const float r = rsqrtf(tot / (float)D_ + 1e-6f);
  const float y0 = a * r * w[tid];
  const float y1 = b * r * w[tid + 256];
  if (outb) {
    outb[(size_t)bt * D_ + tid] = f2bf(y0);
    outb[(size_t)bt * D_ + tid + 256] = f2bf(y1);
  }
  if (outf) {
    outf[(size_t)bt * D_ + tid] = y0;
    outf[(size_t)bt * D_ + tid + 256] = y1;
  }
}

// RoPE + bf16 convert (do_rope=1) or plain scale+convert (do_rope=0)
__global__ void rope_conv_k(const float* __restrict__ in,
                            __bf16* __restrict__ outb, float scale,
                            int do_rope) {
  const int bt = blockIdx.x;
  const int t = bt % T_;
  const float* row = in + (size_t)bt * D_;
  __bf16* orow = outb + (size_t)bt * D_;
  const int tid = threadIdx.x;
  if (do_rope) {
    const int h = tid >> 5, d = tid & 31;
    const int c1 = h * HD_ + d, c2 = c1 + 32;
    const float f = (float)t * __powf(10000.0f, -(float)d / 32.0f);
    const float cs = __cosf(f), sn = __sinf(f);
    const float x1 = row[c1], x2 = row[c2];
    orow[c1] = f2bf((x1 * cs - x2 * sn) * scale);
    orow[c2] = f2bf((x2 * cs + x1 * sn) * scale);
  } else {
    orow[tid] = f2bf(row[tid] * scale);
    orow[tid + 256] = f2bf(row[tid + 256] * scale);
  }
}

// fp32 (K,N) row-major -> bf16 (N,K) row-major
__global__ void wconv_k(const float* __restrict__ in, __bf16* __restrict__ out,
                        int K, int N) {
  const size_t i = (size_t)blockIdx.x * blockDim.x + threadIdx.x;
  if (i >= (size_t)K * N) return;
  const int k = (int)(i / (size_t)N);
  const int n = (int)(i % (size_t)N);
  out[(size_t)n * K + k] = f2bf(in[i]);
}

__device__ __forceinline__ bool is_sep_c(int c) {
  const char s[] = " \t\n\r.,;:!?()[]{}\"'+-*/=<>|&^~%@#$\\";
  for (int i = 0; s[i]; ++i)
    if (c == (int)(unsigned char)s[i]) return true;
  return false;
}

__global__ void seg_scan_k(const int* __restrict__ tok, int* __restrict__ seg) {
  const int b = blockIdx.x * blockDim.x + threadIdx.x;
  if (b >= B_) return;
  int acc = 0;
  for (int t = 0; t < T_; ++t) {
    const int c = tok[b * T_ + t] & 255;
    acc += is_sep_c(c) ? 1 : 0;
    seg[b * T_ + t] = acc;
  }
}

__global__ void zero_k(float* __restrict__ p, size_t n) {
  const size_t i = (size_t)blockIdx.x * blockDim.x + threadIdx.x;
  if (i < n) p[i] = 0.0f;
}

__global__ void seg_acc_k(const float* __restrict__ xn,
                          const int* __restrict__ seg,
                          float* __restrict__ ssum, float* __restrict__ scnt) {
  const int bt = blockIdx.x;
  const int tid = threadIdx.x;
  const int b = bt / T_;
  const int s = seg[bt] + b * (T_ + 1);
  atomicAdd(&ssum[(size_t)s * D_ + tid], xn[(size_t)bt * D_ + tid]);
  atomicAdd(&ssum[(size_t)s * D_ + tid + 256], xn[(size_t)bt * D_ + tid + 256]);
  if (tid == 0) atomicAdd(&scnt[s], 1.0f);
}

__global__ void combine_k(const float* __restrict__ xn,
                          const int* __restrict__ seg,
                          const float* __restrict__ ssum,
                          const float* __restrict__ scnt,
                          float* __restrict__ out) {
  const int bt = blockIdx.x;
  const int tid = threadIdx.x;
  const int b = bt / T_;
  const int s = seg[bt] + b * (T_ + 1);
  const float inv = 1.0f / fmaxf(scnt[s], 1.0f);
#pragma unroll
  for (int q = 0; q < 2; ++q) {
    const int c = tid + q * 256;
    out[(size_t)bt * D_ + c] =
        0.5f * xn[(size_t)bt * D_ + c] + 0.5f * ssum[(size_t)s * D_ + c] * inv;
  }
}

// ---------------------------------------------------------------------------
// launch
// ---------------------------------------------------------------------------
extern "C" void kernel_launch(void* const* d_in, const int* in_sizes, int n_in,
                              void* d_out, int out_size, void* d_ws,
                              size_t ws_size, hipStream_t stream) {
  const int* tokens = (const int*)d_in[0];
  const float* embed_table = (const float*)d_in[1];
  const float* attn_norm_w = (const float*)d_in[2];
  const float* wq = (const float*)d_in[3];
  const float* wk = (const float*)d_in[4];
  const float* wv = (const float*)d_in[5];
  const float* wo = (const float*)d_in[6];
  const float* ffn_norm_w = (const float*)d_in[7];
  const float* w1 = (const float*)d_in[8];
  const float* w2 = (const float*)d_in[9];
  const float* w3 = (const float*)d_in[10];
  const float* final_norm_w = (const float*)d_in[11];
  float* out = (float*)d_out;

  // ---- workspace carve ----
  char* ws = (char*)d_ws;
  size_t off = 0;
  auto carve = [&](size_t bytes) -> void* {
    void* p = ws + off;
    off += (bytes + 255) & ~(size_t)255;
    return p;
  };
  float* x = (float*)carve((size_t)BT_ * D_ * 4);
  float* xn = (float*)carve((size_t)BT_ * D_ * 4);
  __bf16* hb = (__bf16*)carve((size_t)BT_ * D_ * 2);
  float* bigA = (float*)carve((size_t)BT_ * FF_ * 4);   // q|k|v (attn) or u (ffn)
  __bf16* bigB = (__bf16*)carve((size_t)BT_ * FF_ * 2); // qb|kb|vb|ob or g
  const size_t WPL = 4 * (size_t)D_ * D_ + 3 * (size_t)D_ * FF_;
  __bf16* wts = (__bf16*)carve(L_ * WPL * 2);
  float* ssum = (float*)carve((size_t)NSEG_ * D_ * 4);
  float* scnt = (float*)carve((size_t)NSEG_ * 4);
  int* seg = (int*)carve((size_t)BT_ * 4);

  float* qf = bigA;
  float* kf = bigA + (size_t)BT_ * D_;
  float* vf = bigA + 2 * (size_t)BT_ * D_;
  __bf16* qb = bigB;
  __bf16* kb = bigB + (size_t)BT_ * D_;
  __bf16* vb = bigB + 2 * (size_t)BT_ * D_;
  __bf16* ob = bigB + 3 * (size_t)BT_ * D_;
  float* u = bigA;     // FFN phase reuse
  __bf16* g = bigB;

  auto gemm = [&](const __bf16* A, const __bf16* Wt, float* oF, __bf16* oB,
                  const float* aux, int M, int N, int K, int mode) {
    const int waves = (M / 16) * (N / 64);
    const int blocks = (waves + 7) / 8;
    gemm_bf16_wmma<<<blocks, 256, 0, stream>>>(A, Wt, oF, oB, aux, M, N, K, mode);
  };
  auto wconv = [&](const float* in, __bf16* o, int K, int N) {
    const size_t n = (size_t)K * N;
    wconv_k<<<(unsigned)((n + 255) / 256), 256, 0, stream>>>(in, o, K, N);
  };

  // ---- one-time weight transpose + bf16 convert ----
  for (int l = 0; l < L_; ++l) {
    __bf16* wb = wts + (size_t)l * WPL;
    __bf16* wqt = wb;
    __bf16* wkt = wqt + (size_t)D_ * D_;
    __bf16* wvt = wkt + (size_t)D_ * D_;
    __bf16* wot = wvt + (size_t)D_ * D_;
    __bf16* w1t = wot + (size_t)D_ * D_;
    __bf16* w2t = w1t + (size_t)D_ * FF_;
    __bf16* w3t = w2t + (size_t)D_ * FF_;
    wconv(wq + (size_t)l * D_ * D_, wqt, D_, D_);
    wconv(wk + (size_t)l * D_ * D_, wkt, D_, D_);
    wconv(wv + (size_t)l * D_ * D_, wvt, D_, D_);
    wconv(wo + (size_t)l * D_ * D_, wot, D_, D_);
    wconv(w1 + (size_t)l * D_ * FF_, w1t, D_, FF_);
    wconv(w2 + (size_t)l * D_ * FF_, w2t, D_, FF_);
    wconv(w3 + (size_t)l * FF_ * D_, w3t, FF_, D_);
  }

  // ---- embed ----
  embed_k<<<BT_, 256, 0, stream>>>(tokens, embed_table, x);

  // ---- layers ----
  for (int l = 0; l < L_; ++l) {
    __bf16* wb = wts + (size_t)l * WPL;
    __bf16* wqt = wb;
    __bf16* wkt = wqt + (size_t)D_ * D_;
    __bf16* wvt = wkt + (size_t)D_ * D_;
    __bf16* wot = wvt + (size_t)D_ * D_;
    __bf16* w1t = wot + (size_t)D_ * D_;
    __bf16* w2t = w1t + (size_t)D_ * FF_;
    __bf16* w3t = w2t + (size_t)D_ * FF_;

    rmsnorm_k<<<BT_, 256, 0, stream>>>(x, attn_norm_w + l * D_, hb, nullptr);
    gemm(hb, wqt, qf, nullptr, nullptr, BT_, D_, D_, 0);
    gemm(hb, wkt, kf, nullptr, nullptr, BT_, D_, D_, 0);
    gemm(hb, wvt, vf, nullptr, nullptr, BT_, D_, D_, 0);

    rope_conv_k<<<BT_, 256, 0, stream>>>(qf, qb, 0.125f, 1);  // fold 1/sqrt(HD)
    rope_conv_k<<<BT_, 256, 0, stream>>>(kf, kb, 1.0f, 1);
    rope_conv_k<<<BT_, 256, 0, stream>>>(vf, vb, 1.0f, 0);

    attn_wmma<<<dim3(B_ * H_, T_ / (ATT_WAVES * 16)), ATT_WAVES * 32, 0,
                stream>>>(qb, kb, vb, ob);

    gemm(ob, wot, x, nullptr, x, BT_, D_, D_, 1);             // x += o @ wo

    rmsnorm_k<<<BT_, 256, 0, stream>>>(x, ffn_norm_w + l * D_, hb, nullptr);
    gemm(hb, w1t, u, nullptr, nullptr, BT_, FF_, D_, 0);      // u = h @ w1
    gemm(hb, w2t, nullptr, g, u, BT_, FF_, D_, 2);            // g = silu(u)*(h@w2)
    gemm(g, w3t, x, nullptr, x, BT_, D_, FF_, 1);             // x += g @ w3
  }

  // ---- final norm + segment mean ----
  rmsnorm_k<<<BT_, 256, 0, stream>>>(x, final_norm_w, nullptr, xn);
  seg_scan_k<<<1, 32, 0, stream>>>(tokens, seg);
  {
    const size_t n = (size_t)NSEG_ * D_;
    zero_k<<<(unsigned)((n + 255) / 256), 256, 0, stream>>>(ssum, n);
    zero_k<<<(NSEG_ + 255) / 256, 256, 0, stream>>>(scnt, (size_t)NSEG_);
  }
  seg_acc_k<<<BT_, 256, 0, stream>>>(xn, seg, ssum, scnt);
  combine_k<<<BT_, 256, 0, stream>>>(xn, seg, ssum, scnt, out);
}
